// Strategy_Layer_62551903699450
// MI455X (gfx1250) — matrix operand order repaired and checked by reading, not compiled, not run on hardware
//
#include <hip/hip_runtime.h>
#include <hip/hip_bf16.h>

// ---------------------------------------------------------------------------
// GRU single step, B=32768, D=128, Hd=2048, O=1  — MI455X (gfx1250, wave32)
// Two fused BF16-WMMA GEMM kernels, double-buffered LDS pipeline + prefetch.
// ---------------------------------------------------------------------------

typedef __attribute__((ext_vector_type(16))) __bf16 v16bf;
typedef __attribute__((ext_vector_type(8)))  float  v8f;

#define B_SZ 32768
#define D_SZ 128
#define HD   2048
#define BM   128
#define BN   128
#define KC   32
#define KIT_X (D_SZ / KC)          // 4 x-phase iterations
#define KIT_H (HD / KC)            // 64 H-phase iterations
#define NIT   (KIT_X + KIT_H)      // 68 total K-steps

// ---- bf16 helpers ---------------------------------------------------------
__device__ __forceinline__ unsigned int pack_bf16x2(float lo, float hi) {
  __hip_bfloat162 h = __float22bfloat162_rn(make_float2(lo, hi));  // x=lo (low 16b)
  unsigned int u;
  __builtin_memcpy(&u, &h, 4);
  return u;
}
__device__ __forceinline__ unsigned short f32_to_bf16(float f) {
  unsigned int u = __float_as_uint(f);
  u = (u + 0x7FFFu + ((u >> 16) & 1u)) >> 16;
  return (unsigned short)u;
}
__device__ __forceinline__ float bf16_to_f32(unsigned short h) {
  return __uint_as_float(((unsigned int)h) << 16);
}

// ---------------------------------------------------------------------------
// WMMA fragment LDS layout (ISA 7.12.2):
// A (16-bit, 16x32 sub-tile): lane L<16: row M=L, K={0..7,16..23};
//   lane L>=16: row M=L-16, K={8..15,24..31}. 32B contiguous per lane.
// B (16-bit, 32x16 sub-tile): lane L<16: col N=L, K={0..7,16..23}; lane L>=16:
//   col N=L-16, K={8..15,24..31}. 32B contiguous per lane. Sub-tile = 1KB.
// ---------------------------------------------------------------------------

// ---- staging: global -> packed-bf16 VGPRs (conversion at load time) -------
__device__ __forceinline__ void load_a_f32_packed(uint2* r, const float* __restrict__ src,
                                                  int lda, int rowbase, int kb, int tid) {
#pragma unroll
  for (int i = 0; i < 4; ++i) {
    int q  = i * 256 + tid;              // 1024 float4 groups, coalesced
    int rr = q >> 3;
    int k  = (q & 7) << 2;
    const float* p = src + (size_t)(rowbase + rr) * lda + kb + k;
    if (i == 0) __builtin_prefetch(p + 2 * KC, 0, 0);   // A-stream, 2 K-steps ahead
    float4 v = *(const float4*)p;
    r[i].x = pack_bf16x2(v.x, v.y);
    r[i].y = pack_bf16x2(v.z, v.w);
  }
}
__device__ __forceinline__ void load_a_bf16_regs(uint4* r, const unsigned short* __restrict__ src,
                                                 int lda, int rowbase, int kb, int tid) {
#pragma unroll
  for (int i = 0; i < 2; ++i) {
    int q  = i * 256 + tid;              // 512 groups of 8 bf16, coalesced
    int rr = q >> 2;
    int k  = (q & 3) << 3;
    const unsigned short* p = src + (size_t)(rowbase + rr) * lda + kb + k;
    if (i == 0) __builtin_prefetch(p + 2 * KC, 0, 0);
    r[i] = *(const uint4*)p;
  }
}
__device__ __forceinline__ void load_w_packed(unsigned int* pk, const float* __restrict__ W,
                                              int kb, int nbase, int tid) {
  int n = tid & 127, khalf = tid >> 7;
  const float* col = W + (size_t)(kb + khalf * 8) * HD + nbase + n;  // lane-coalesced over n
  __builtin_prefetch(col + (size_t)(2 * KC) * HD, 0, 0);              // W-stream, 2 K-steps ahead
#pragma unroll
  for (int j = 0; j < 8; ++j)
    pk[j] = pack_bf16x2(col[(2 * j + (j >= 4 ? 8 : 0)) * HD], col[(2 * j + 1 + (j >= 4 ? 8 : 0)) * HD]);
}

// ---- LDS commit (pure stores, fragment layout) ----------------------------
__device__ __forceinline__ void write_a_lds_f32(unsigned char* ldsA, const uint2* r, int tid) {
#pragma unroll
  for (int i = 0; i < 4; ++i) {
    int q  = i * 256 + tid;
    int rr = q >> 3;
    int k  = (q & 7) << 2;
    int lane = (rr & 15) + (((k & 15) >= 8) ? 16 : 0);
    int byte = ((k >> 4) << 4) + ((k & 7) << 1);
    *(uint2*)(ldsA + ((rr >> 4) << 10) + (lane << 5) + byte) = r[i];
  }
}
__device__ __forceinline__ void write_a_lds_bf16(unsigned char* ldsA, const uint4* r, int tid) {
#pragma unroll
  for (int i = 0; i < 2; ++i) {
    int q  = i * 256 + tid;
    int rr = q >> 2;
    int k  = (q & 3) << 3;
    int lane = (rr & 15) + (((k & 15) >= 8) ? 16 : 0);
    int byte = (k >> 4) << 4;
    *(uint4*)(ldsA + ((rr >> 4) << 10) + (lane << 5) + byte) = r[i];
  }
}
__device__ __forceinline__ void write_w_lds(unsigned char* ldsB, const unsigned int* pk, int tid) {
  int n = tid & 127, khalf = tid >> 7;
  int lane = (n & 15) + (khalf << 4);
  uint4* dst = (uint4*)(ldsB + ((n >> 4) << 10) + (lane << 5));
  dst[0] = make_uint4(pk[0], pk[1], pk[2], pk[3]);
  dst[1] = make_uint4(pk[4], pk[5], pk[6], pk[7]);
}

__device__ __forceinline__ v16bf load_frag(const unsigned char* lds, int subtile, int lane) {
  return *(const v16bf*)(lds + (subtile << 10) + (lane << 5));
}
__device__ __forceinline__ float sigmoidf_fast(float x) {
  return 1.0f / (1.0f + __expf(-x));
}

// ---------------------------------------------------------------------------
// Kernel 1: Z = sigmoid(x@W_xz + H@W_hz + b_z), R = sigmoid(x@W_xr + H@W_hr + b_r)
// writes Zws (bf16) and RHws = R*H (bf16). Double-buffered LDS pipeline.
// ---------------------------------------------------------------------------
__global__ __launch_bounds__(256) void gru_gates_zr(
    const float* __restrict__ x, const float* __restrict__ H,
    const float* __restrict__ W_xz, const float* __restrict__ W_hz, const float* __restrict__ b_z,
    const float* __restrict__ W_xr, const float* __restrict__ W_hr, const float* __restrict__ b_r,
    unsigned short* __restrict__ Zws, unsigned short* __restrict__ RHws) {
  __shared__ __align__(32) unsigned char smem[2 * 24576];   // [buf][A 8K | Bz 8K | Br 8K]

  const int tid  = threadIdx.x;
  const int lane = tid & 31;
  const int w    = tid >> 5;
  const int wm   = w & 3;             // 4 waves along M
  const int wn   = w >> 2;            // 2 waves along N
  const int rowbase = blockIdx.y * BM;
  const int nbase   = blockIdx.x * BN;

  v8f zero = {};
  v8f accZ[2][4], accR[2][4];
#pragma unroll
  for (int mt = 0; mt < 2; ++mt)
#pragma unroll
    for (int nt = 0; nt < 4; ++nt) { accZ[mt][nt] = zero; accR[mt][nt] = zero; }

  auto desc = [&](int it, const float*& A, int& lda, int& kb, const float*& Wz, const float*& Wr) {
    if (it < KIT_X) { A = x; lda = D_SZ; kb = it * KC;           Wz = W_xz; Wr = W_xr; }
    else            { A = H; lda = HD;   kb = (it - KIT_X) * KC; Wz = W_hz; Wr = W_hr; }
  };

  uint2        ra[4];
  unsigned int rz[8], rr_[8];

  { // prologue: stage iteration 0 into LDS buffer 0
    const float *A, *Wz_, *Wr_; int lda, kb;
    desc(0, A, lda, kb, Wz_, Wr_);
    load_a_f32_packed(ra, A, lda, rowbase, kb, tid);
    load_w_packed(rz,  Wz_, kb, nbase, tid);
    load_w_packed(rr_, Wr_, kb, nbase, tid);
    write_a_lds_f32(smem, ra, tid);
    write_w_lds(smem + 8192, rz, tid);
    write_w_lds(smem + 16384, rr_, tid);
  }
  __syncthreads();

  for (int it = 0; it < NIT; ++it) {
    const int buf = it & 1;
    unsigned char* cur = smem + buf * 24576;
    const bool more = (it + 1 < NIT);

    // issue next iteration's global loads + convert (co-executes with WMMA)
    if (more) {
      const float *A, *Wz_, *Wr_; int lda, kb;
      desc(it + 1, A, lda, kb, Wz_, Wr_);
      load_a_f32_packed(ra, A, lda, rowbase, kb, tid);
      load_w_packed(rz,  Wz_, kb, nbase, tid);
      load_w_packed(rr_, Wr_, kb, nbase, tid);
    }

    // compute on current buffer (gate-sequential to limit live fragments)
    v16bf af[2];
#pragma unroll
    for (int mt = 0; mt < 2; ++mt) af[mt] = load_frag(cur, wm * 2 + mt, lane);
    {
      v16bf bzf[4];
#pragma unroll
      for (int nt = 0; nt < 4; ++nt) bzf[nt] = load_frag(cur + 8192, wn * 4 + nt, lane);
#pragma unroll
      for (int mt = 0; mt < 2; ++mt)
#pragma unroll
        for (int nt = 0; nt < 4; ++nt)
          accZ[mt][nt] = __builtin_amdgcn_wmma_f32_16x16x32_bf16(
              false, af[mt], false, bzf[nt], (short)0, accZ[mt][nt], false, false);
    }
    {
      v16bf brf[4];
#pragma unroll
      for (int nt = 0; nt < 4; ++nt) brf[nt] = load_frag(cur + 16384, wn * 4 + nt, lane);
#pragma unroll
      for (int mt = 0; mt < 2; ++mt)
#pragma unroll
        for (int nt = 0; nt < 4; ++nt)
          accR[mt][nt] = __builtin_amdgcn_wmma_f32_16x16x32_bf16(
              false, af[mt], false, brf[nt], (short)0, accR[mt][nt], false, false);
    }

    // commit next iteration into the other buffer
    if (more) {
      unsigned char* nxt = smem + (buf ^ 1) * 24576;
      write_a_lds_f32(nxt, ra, tid);
      write_w_lds(nxt + 8192, rz, tid);
      write_w_lds(nxt + 16384, rr_, tid);
    }
    __syncthreads();
  }

  // Epilogue: bias + sigmoid, emit Z and R*H as bf16.
  // C layout: VGPR v -> (M = v + 8*(lane>=16), N = lane%16).
  const int moff = (lane >> 4) << 3;
  const int ncl  = lane & 15;
#pragma unroll
  for (int mt = 0; mt < 2; ++mt) {
    const int mb = rowbase + (wm * 2 + mt) * 16 + moff;
#pragma unroll
    for (int nt = 0; nt < 4; ++nt) {
      const int col  = nbase + (wn * 4 + nt) * 16 + ncl;
      const float bzv = b_z[col];
      const float brv = b_r[col];
#pragma unroll
      for (int v = 0; v < 8; ++v) {
        const size_t idx = (size_t)(mb + v) * HD + col;
        float z = sigmoidf_fast(accZ[mt][nt][v] + bzv);
        float r = sigmoidf_fast(accR[mt][nt][v] + brv);
        float h = H[idx];
        Zws[idx]  = f32_to_bf16(z);
        RHws[idx] = f32_to_bf16(r * h);
      }
    }
  }
}

// ---------------------------------------------------------------------------
// Kernel 2: H_tilda = tanh(x@W_xh + RH@W_hh + b_h); H_new = Z*H + (1-Z)*H_tilda;
// Y += H_new @ W_hq (per-row dot fused via lane reduction + atomicAdd).
// ---------------------------------------------------------------------------
__global__ __launch_bounds__(256) void gru_cand_out(
    const float* __restrict__ x, const float* __restrict__ H,
    const float* __restrict__ W_xh, const float* __restrict__ W_hh, const float* __restrict__ b_h,
    const float* __restrict__ W_hq,
    const unsigned short* __restrict__ Zws, const unsigned short* __restrict__ RHws,
    float* __restrict__ Y) {
  __shared__ __align__(32) unsigned char smem[2 * 16384];   // [buf][A 8K | B 8K]

  const int tid  = threadIdx.x;
  const int lane = tid & 31;
  const int w    = tid >> 5;
  const int wm   = w & 3;
  const int wn   = w >> 2;
  const int rowbase = blockIdx.y * BM;
  const int nbase   = blockIdx.x * BN;

  v8f zero = {};
  v8f acc[2][4];
#pragma unroll
  for (int mt = 0; mt < 2; ++mt)
#pragma unroll
    for (int nt = 0; nt < 4; ++nt) acc[mt][nt] = zero;

  uint2        raf[4];
  uint4        rab[2];
  unsigned int rw[8];

  auto load_iter = [&](int it) {
    if (it < KIT_X) {
      load_a_f32_packed(raf, x, D_SZ, rowbase, it * KC, tid);
      load_w_packed(rw, W_xh, it * KC, nbase, tid);
    } else {
      int kb = (it - KIT_X) * KC;
      load_a_bf16_regs(rab, RHws, HD, rowbase, kb, tid);
      load_w_packed(rw, W_hh, kb, nbase, tid);
    }
  };
  auto write_iter = [&](int it, unsigned char* b) {
    if (it < KIT_X) write_a_lds_f32(b, raf, tid);
    else            write_a_lds_bf16(b, rab, tid);
    write_w_lds(b + 8192, rw, tid);
  };

  load_iter(0);
  write_iter(0, smem);
  __syncthreads();

  for (int it = 0; it < NIT; ++it) {
    const int buf = it & 1;
    unsigned char* cur = smem + buf * 16384;
    const bool more = (it + 1 < NIT);

    if (more) load_iter(it + 1);

    v16bf af[2], bf[4];
#pragma unroll
    for (int mt = 0; mt < 2; ++mt) af[mt] = load_frag(cur, wm * 2 + mt, lane);
#pragma unroll
    for (int nt = 0; nt < 4; ++nt) bf[nt] = load_frag(cur + 8192, wn * 4 + nt, lane);
#pragma unroll
    for (int mt = 0; mt < 2; ++mt)
#pragma unroll
      for (int nt = 0; nt < 4; ++nt)
        acc[mt][nt] = __builtin_amdgcn_wmma_f32_16x16x32_bf16(
            false, af[mt], false, bf[nt], (short)0, acc[mt][nt], false, false);

    if (more) write_iter(it + 1, smem + (buf ^ 1) * 16384);
    __syncthreads();
  }

  // Epilogue: tanh, blend, fused Hd->1 projection.
  const int moff = (lane >> 4) << 3;
  const int ncl  = lane & 15;
  float ysum[2][8];
#pragma unroll
  for (int mt = 0; mt < 2; ++mt)
#pragma unroll
    for (int v = 0; v < 8; ++v) ysum[mt][v] = 0.0f;

#pragma unroll
  for (int mt = 0; mt < 2; ++mt) {
    const int mb = rowbase + (wm * 2 + mt) * 16 + moff;
#pragma unroll
    for (int nt = 0; nt < 4; ++nt) {
      const int col  = nbase + (wn * 4 + nt) * 16 + ncl;
      const float bhv = b_h[col];
      const float wqv = W_hq[col];
#pragma unroll
      for (int v = 0; v < 8; ++v) {
        const size_t idx = (size_t)(mb + v) * HD + col;
        float ht = tanhf(acc[mt][nt][v] + bhv);
        float z  = bf16_to_f32(Zws[idx]);
        float h  = H[idx];
        float hn = z * h + (1.0f - z) * ht;
        ysum[mt][v] += hn * wqv;
      }
    }
  }

  // Cross-lane reduction over N (16 lanes per half-wave; wave32).
#pragma unroll
  for (int mt = 0; mt < 2; ++mt) {
    const int mb = rowbase + (wm * 2 + mt) * 16 + moff;
#pragma unroll
    for (int v = 0; v < 8; ++v) {
      float s = ysum[mt][v];
      s += __shfl_xor(s, 1, 16);
      s += __shfl_xor(s, 2, 16);
      s += __shfl_xor(s, 4, 16);
      s += __shfl_xor(s, 8, 16);
      if ((lane & 15) == 0) atomicAdd(&Y[mb + v], s);
    }
  }
}

__global__ void init_y(float* __restrict__ Y, const float* __restrict__ b_q) {
  int i = blockIdx.x * blockDim.x + threadIdx.x;
  if (i < B_SZ) Y[i] = b_q[0];
}

// ---------------------------------------------------------------------------
extern "C" void kernel_launch(void* const* d_in, const int* in_sizes, int n_in,
                              void* d_out, int out_size, void* d_ws, size_t ws_size,
                              hipStream_t stream) {
  const float* x    = (const float*)d_in[0];
  const float* H    = (const float*)d_in[1];
  const float* W_xz = (const float*)d_in[2];
  const float* W_hz = (const float*)d_in[3];
  const float* b_z  = (const float*)d_in[4];
  const float* W_xr = (const float*)d_in[5];
  const float* W_hr = (const float*)d_in[6];
  const float* b_r  = (const float*)d_in[7];
  const float* W_xh = (const float*)d_in[8];
  const float* W_hh = (const float*)d_in[9];
  const float* b_h  = (const float*)d_in[10];
  const float* W_hq = (const float*)d_in[11];
  const float* b_q  = (const float*)d_in[12];
  float* Y = (float*)d_out;

  unsigned short* Zws  = (unsigned short*)d_ws;                 // B*HD bf16
  unsigned short* RHws = Zws + (size_t)B_SZ * HD;               // B*HD bf16

  dim3 grid(HD / BN, B_SZ / BM);   // 16 x 256 blocks, 256 threads (8 wave32)
  gru_gates_zr<<<grid, 256, 0, stream>>>(x, H, W_xz, W_hz, b_z, W_xr, W_hr, b_r, Zws, RHws);
  init_y<<<(B_SZ + 255) / 256, 256, 0, stream>>>(Y, b_q);
  gru_cand_out<<<grid, 256, 0, stream>>>(x, H, W_xh, W_hh, b_h, W_hq, Zws, RHws, Y);
}
